// GPT_74577812127945
// MI455X (gfx1250) — matrix operand order repaired and checked by reading, not compiled, not run on hardware
//
#include <hip/hip_runtime.h>
#include <hip/hip_bf16.h>

// ---------------------------------------------------------------------------
// GPT-2-style forward pass for gfx1250 (MI455X).
// GEMMs + attention matmuls via v_wmma_f32_16x16x32_f16 (wave32 WMMA),
// A-tile staging via Tensor Data Mover (tensor_load_to_lds) when available.
// ---------------------------------------------------------------------------

typedef __attribute__((ext_vector_type(16))) _Float16 v16h;
typedef __attribute__((ext_vector_type(8)))  float    v8f;
typedef __attribute__((ext_vector_type(4)))  unsigned int u32x4;
typedef __attribute__((ext_vector_type(8)))  int          i32x8;
typedef __attribute__((ext_vector_type(4)))  int          i32x4;

union FragA { v16h v; _Float16 h[16]; };
union Acc   { v8f  v; float    f[8];  };

#define CEMB   1024
#define TSEQ   1024
#define BB     4
#define MTOT   4096          // B*T rows
#define NHEAD  16
#define HD     64
#define VOCAB  50304
#define NLAYER 6

#if __has_builtin(__builtin_amdgcn_tensor_load_to_lds) && \
    __has_builtin(__builtin_amdgcn_s_wait_tensorcnt)
#define HAVE_TDM 1
#endif

#ifdef HAVE_TDM
// Issue one TDM 2-D tile load: rows x 32 halves, row stride rowLen halves,
// into LDS at lds_byte_off with padded row stride 40 halves (pad 16B / 64B).
__device__ __forceinline__ void tdm_load_tile(const _Float16* gsrc,
                                              unsigned lds_byte_off,
                                              int rowLen, int rows) {
    unsigned long long ga = (unsigned long long)(uintptr_t)gsrc;
    u32x4 g0;
    g0[0] = 1u;                                   // count=1, user mode
    g0[1] = lds_byte_off;                         // lds_addr
    g0[2] = (unsigned)(ga & 0xFFFFFFFFull);       // global_addr[31:0]
    g0[3] = (unsigned)((ga >> 32) & 0x01FFFFFFull) | (2u << 30);  // addr hi | type=2
    i32x8 g1;
    // data_size=2B(1), pad_enable, pad_interval=16 DW(3), pad_amount=4 DW(3)
    g1[0] = (1 << 16) | (1 << 20) | (3 << 22) | (3 << 25);
    g1[1] = (rowLen & 0xFFFF) << 16;              // tensor_dim0 lo16 @ [63:48]
    g1[2] = ((rowLen >> 16) & 0xFFFF) | ((rows & 0xFFFF) << 16); // dim0 hi | dim1 lo
    g1[3] = ((rows >> 16) & 0xFFFF) | (32 << 16); // dim1 hi | tile_dim0=32
    g1[4] = rows;                                 // tile_dim1 (tile_dim2=0)
    g1[5] = rowLen;                               // tensor_dim0_stride lo32
    g1[6] = 0;
    g1[7] = 0;
    i32x4 z4 = {0, 0, 0, 0};
#if __clang_major__ >= 23
    i32x8 z8 = {0, 0, 0, 0, 0, 0, 0, 0};
    __builtin_amdgcn_tensor_load_to_lds(g0, g1, z4, z4, z8, 0);
#else
    __builtin_amdgcn_tensor_load_to_lds(g0, g1, z4, z4, 0);
#endif
}
#endif

// ---------------------------------------------------------------------------
// f32 -> f16 conversion (weights)
// ---------------------------------------------------------------------------
__global__ __launch_bounds__(256) void cvt_f32_f16(const float* __restrict__ s,
                                                   _Float16* __restrict__ d,
                                                   long long n) {
    long long i = (long long)blockIdx.x * blockDim.x + threadIdx.x;
    long long stride = (long long)gridDim.x * blockDim.x;
    for (; i < n; i += stride) d[i] = (_Float16)s[i];
}

// ---------------------------------------------------------------------------
// Embedding: x[row,:] = wte[idx[row],:] + wpe[row % T,:]
// ---------------------------------------------------------------------------
__global__ __launch_bounds__(256) void embed_k(const int* __restrict__ idx,
                                               const float* __restrict__ wte,
                                               const float* __restrict__ wpe,
                                               float* __restrict__ x) {
    const int row = blockIdx.x;
    const int tid = threadIdx.x;
    const int tok = idx[row];
    const int t   = row & (TSEQ - 1);
    float4 a = *(const float4*)(wte + (size_t)tok * CEMB + tid * 4);
    float4 p = *(const float4*)(wpe + (size_t)t   * CEMB + tid * 4);
    float4 o; o.x = a.x + p.x; o.y = a.y + p.y; o.z = a.z + p.z; o.w = a.w + p.w;
    *(float4*)(x + (size_t)row * CEMB + tid * 4) = o;
}

// ---------------------------------------------------------------------------
// LayerNorm: f32 row -> f16 row
// ---------------------------------------------------------------------------
__global__ __launch_bounds__(256) void layernorm_k(const float* __restrict__ x,
                                                   const float* __restrict__ w,
                                                   const float* __restrict__ b,
                                                   _Float16* __restrict__ out) {
    const int row = blockIdx.x, tid = threadIdx.x;
    const float* xr = x + (size_t)row * CEMB;
    float4 v = *(const float4*)(xr + tid * 4);
    float s  = v.x + v.y + v.z + v.w;
    float s2 = v.x * v.x + v.y * v.y + v.z * v.z + v.w * v.w;
    __shared__ float rs[256], rq[256];
    rs[tid] = s; rq[tid] = s2;
    __syncthreads();
    for (int st = 128; st > 0; st >>= 1) {
        if (tid < st) { rs[tid] += rs[tid + st]; rq[tid] += rq[tid + st]; }
        __syncthreads();
    }
    const float mu  = rs[0] * (1.0f / CEMB);
    const float var = rq[0] * (1.0f / CEMB) - mu * mu;
    const float inv = rsqrtf(var + 1e-5f);
    float e[4] = {v.x, v.y, v.z, v.w};
#pragma unroll
    for (int k = 0; k < 4; ++k) {
        const int c = tid * 4 + k;
        out[(size_t)row * CEMB + c] = (_Float16)((e[k] - mu) * inv * w[c] + b[c]);
    }
}

// ---------------------------------------------------------------------------
// Tiled WMMA GEMM:  C[M,N] = A[M,K](f16) * B(f16) + bias, epilogue variants.
//   B_IS_NK=false: B stored [K,N] row-major (weights y = x@W)
//   B_IS_NK=true : B stored [N,K] row-major (tied lm head: wte)
// Block = 256 thr (8 waves, 4x2), tile 256x128, K-step 32, double-buffered
// LDS with single barrier per step; A-tile (and NK B-tile) staged via TDM.
// Each wave: 64x64 = 4x4 WMMA tiles of 16x16.
// ---------------------------------------------------------------------------
#define EPI_F32   0
#define EPI_F16   1
#define EPI_GELU  2
#define EPI_RES   3

#define ASTRIDE 40               // padded LDS row stride in halves (80 B)
#define ABUF    (256 * ASTRIDE)  // 10240 halves per A buffer
#define BBUF    (128 * ASTRIDE)  // 5120 halves per B buffer
#define BOFF    (2 * ABUF)       // B region starts after both A buffers

template <int EPI, bool B_IS_NK>
__global__ __launch_bounds__(256)
void gemm_wmma(const _Float16* __restrict__ A, const _Float16* __restrict__ Bm,
               const float* __restrict__ bias, const float* __restrict__ res,
               float* __restrict__ outF, _Float16* __restrict__ outH,
               int M, int N, int K) {
    __shared__ __align__(16) _Float16 smem[2 * ABUF + 2 * BBUF];  // 61440 B

    const int tid  = threadIdx.x;
    const int lane = tid & 31;
    const int wid  = tid >> 5;
    const int wm   = wid >> 1;          // 0..3
    const int wn   = wid & 1;           // 0..1
    const int half = lane >> 4;         // 0/1
    const int l15  = lane & 15;
    const int n0   = blockIdx.x * 128;
    const int m0   = blockIdx.y * 256;

    Acc acc[4][4];
#pragma unroll
    for (int mt = 0; mt < 4; ++mt)
#pragma unroll
        for (int nt = 0; nt < 4; ++nt)
#pragma unroll
            for (int i = 0; i < 8; ++i) acc[mt][nt].f[i] = 0.0f;

    const int nstep = K >> 5;

    // ---- staging helpers ---------------------------------------------------
    auto stageA = [&](int k0, int buf) {
#ifdef HAVE_TDM
        if (wid == 0)
            tdm_load_tile(A + (size_t)m0 * K + k0, (unsigned)(buf * ABUF * 2), K, 256);
#else
        _Float16* An = smem + buf * ABUF;
#pragma unroll
        for (int c = 0; c < 4; ++c) {
            const int lin = tid + c * 256;          // 1024 chunks of 8 halves
            const int row = lin >> 2, seg = lin & 3;
            const _Float16* src = A + (size_t)(m0 + row) * K + k0 + seg * 8;
            *(uint4*)&An[row * ASTRIDE + seg * 8] = *(const uint4*)src;
        }
        __builtin_prefetch((const void*)(A + (size_t)(m0 + (tid >> 1)) * K + k0 + 32), 0, 1);
#endif
    };
    auto stageB = [&](int k0, int buf) {
        _Float16* Bn = smem + BOFF + buf * BBUF;
        if constexpr (B_IS_NK) {
#ifdef HAVE_TDM
            if (wid == 0)
                tdm_load_tile(Bm + (size_t)n0 * K + k0,
                              (unsigned)((BOFF + buf * BBUF) * 2), K, 128);
#else
#pragma unroll
            for (int c = 0; c < 2; ++c) {
                const int lin = tid + c * 256;      // 512 chunks of 8 halves
                const int row = lin >> 2, seg = lin & 3;
                const _Float16* src = Bm + (size_t)(n0 + row) * K + k0 + seg * 8;
                *(uint4*)&Bn[row * ASTRIDE + seg * 8] = *(const uint4*)src;
            }
#endif
        } else {
            // transpose-stage [K,N] weights into Bn[n][k]
#pragma unroll
            for (int c = 0; c < 2; ++c) {
                const int lin = tid + c * 256;      // 32 k-rows x 16 n-segs
                const int kr = lin >> 4, ns = lin & 15;
                const _Float16* src = Bm + (size_t)(k0 + kr) * N + n0 + ns * 8;
                uint4 dv = *(const uint4*)src;
                const _Float16* hp = (const _Float16*)&dv;
#pragma unroll
                for (int e = 0; e < 8; ++e) Bn[(ns * 8 + e) * ASTRIDE + kr] = hp[e];
            }
        }
    };
    auto tdm_fence = [&]() {
#ifdef HAVE_TDM
        if (wid == 0) __builtin_amdgcn_s_wait_tensorcnt(0);
#endif
    };

    // ---- pipeline prologue -------------------------------------------------
    stageA(0, 0);
    stageB(0, 0);
    tdm_fence();
    __syncthreads();

    // ---- main pipelined K loop --------------------------------------------
    for (int s = 0; s < nstep; ++s) {
        const int cur = s & 1;
        if (s + 1 < nstep) {
            stageA((s + 1) << 5, cur ^ 1);
            stageB((s + 1) << 5, cur ^ 1);
        }
        const _Float16* Ab = smem + cur * ABUF;
        const _Float16* Bb = smem + BOFF + cur * BBUF;

        // A fragments (lanes0-15: K=0..15, lanes16-31: K=16..31)
        FragA af[4];
#pragma unroll
        for (int mt = 0; mt < 4; ++mt) {
            const int row = wm * 64 + mt * 16 + l15;
            const _Float16* p0 = Ab + row * ASTRIDE + half * 16;
#pragma unroll
            for (int i = 0; i < 8; ++i) { af[mt].h[i] = p0[i]; af[mt].h[8 + i] = p0[8 + i]; }
        }
        // B fragments (lanes0-15: K=0..15, lanes16-31: K=16..31)
        FragA bf[4];
#pragma unroll
        for (int nt = 0; nt < 4; ++nt) {
            const int col = wn * 64 + nt * 16 + l15;
            const _Float16* p = Bb + col * ASTRIDE + half * 16;
#pragma unroll
            for (int i = 0; i < 16; ++i) bf[nt].h[i] = p[i];
        }
#pragma unroll
        for (int mt = 0; mt < 4; ++mt)
#pragma unroll
            for (int nt = 0; nt < 4; ++nt)
                acc[mt][nt].v = __builtin_amdgcn_wmma_f32_16x16x32_f16(
                    false, af[mt].v, false, bf[nt].v, (short)0, acc[mt][nt].v, false, false);

        if (s + 1 < nstep) tdm_fence();
        __syncthreads();
    }

    // ---- epilogue (C layout: VGPR r -> M=r (lanes<16) / r+8 (lanes>=16))
#pragma unroll
    for (int mt = 0; mt < 4; ++mt)
#pragma unroll
        for (int nt = 0; nt < 4; ++nt) {
            const int gcol = n0 + wn * 64 + nt * 16 + l15;
            const float bv = bias ? bias[gcol] : 0.0f;
#pragma unroll
            for (int r = 0; r < 8; ++r) {
                const int grow = m0 + wm * 64 + mt * 16 + half * 8 + r;
                const size_t o = (size_t)grow * N + gcol;
                float v = acc[mt][nt].f[r] + bv;
                if constexpr (EPI == EPI_F32)  outF[o] = v;
                if constexpr (EPI == EPI_F16)  outH[o] = (_Float16)v;
                if constexpr (EPI == EPI_GELU) {
                    float g = 0.5f * v * (1.0f + erff(v * 0.70710678118f));
                    outH[o] = (_Float16)g;
                }
                if constexpr (EPI == EPI_RES)  outF[o] = res[o] + v;
            }
        }
}

// ---------------------------------------------------------------------------
// Flash attention with decay folded into logits:
//   weights = softmax_j<=i( (q.k)/8 + r*j )      [r*i cancels in renorm]
// Block = (query-tile 64, head, batch), 128 threads = 4 waves; wave -> 16 rows.
// ---------------------------------------------------------------------------
__global__ __launch_bounds__(128)
void attn_flash(const _Float16* __restrict__ qkv, const float* __restrict__ decay,
                _Float16* __restrict__ y) {
    const int qb = blockIdx.x, h = blockIdx.y, b = blockIdx.z;
    const int tid = threadIdx.x, lane = tid & 31, w = tid >> 5;
    const int half = lane >> 4, l15 = lane & 15;
    const float rate = decay[h];

    __shared__ __align__(16) _Float16 Ql[64][72];       // stride 144B (16B mult)
    __shared__ __align__(16) _Float16 Kl[64][72];
    __shared__ __align__(16) _Float16 Vt[64][72];       // transposed: [dim][key]
    __shared__ __align__(16) _Float16 Pl[4][16][72];

    const size_t rowQ = (size_t)(b * TSEQ + qb * 64);

    // stage Q (pre-scaled by 1/sqrt(64))
#pragma unroll
    for (int c = 0; c < 4; ++c) {
        const int lin = tid + c * 128;
        const int row = lin >> 3, seg = lin & 7;
        const _Float16* src = qkv + (rowQ + row) * 3072 + h * HD + seg * 8;
#pragma unroll
        for (int e = 0; e < 8; ++e)
            Ql[row][seg * 8 + e] = (_Float16)((float)src[e] * 0.125f);
    }
    __syncthreads();

    FragA qf[2];
#pragma unroll
    for (int ks = 0; ks < 2; ++ks) {
        const int row = w * 16 + l15;
#pragma unroll
        for (int i = 0; i < 8; ++i) {
            qf[ks].h[i]     = Ql[row][ks * 32 + half * 16 + i];
            qf[ks].h[8 + i] = Ql[row][ks * 32 + half * 16 + 8 + i];
        }
    }

    Acc O[4];
    float m_i[8], l_i[8];
#pragma unroll
    for (int dt = 0; dt < 4; ++dt)
#pragma unroll
        for (int i = 0; i < 8; ++i) O[dt].f[i] = 0.0f;
#pragma unroll
    for (int r = 0; r < 8; ++r) { m_i[r] = -1e30f; l_i[r] = 0.0f; }

    for (int kt = 0; kt <= qb; ++kt) {
        const size_t rowK = (size_t)(b * TSEQ + kt * 64);
        // stage K and V^T
#pragma unroll
        for (int c = 0; c < 4; ++c) {
            const int lin = tid + c * 128;
            const int row = lin >> 3, seg = lin & 7;
            const _Float16* ksrc = qkv + (rowK + row) * 3072 + 1024 + h * HD + seg * 8;
            *(uint4*)&Kl[row][seg * 8] = *(const uint4*)ksrc;
            const _Float16* vsrc = qkv + (rowK + row) * 3072 + 2048 + h * HD + seg * 8;
#pragma unroll
            for (int e = 0; e < 8; ++e) Vt[seg * 8 + e][row] = vsrc[e];
        }
        __syncthreads();

        // S = Q * K^T  (4 key subtiles of 16)
        Acc s[4];
#pragma unroll
        for (int st = 0; st < 4; ++st) {
#pragma unroll
            for (int i = 0; i < 8; ++i) s[st].f[i] = 0.0f;
            FragA kf0, kf1;
            const int col = st * 16 + l15;
#pragma unroll
            for (int i = 0; i < 16; ++i) {
                kf0.h[i] = Kl[col][half * 16 + i];
                kf1.h[i] = Kl[col][32 + half * 16 + i];
            }
            s[st].v = __builtin_amdgcn_wmma_f32_16x16x32_f16(
                false, qf[0].v, false, kf0.v, (short)0, s[st].v, false, false);
            s[st].v = __builtin_amdgcn_wmma_f32_16x16x32_f16(
                false, qf[1].v, false, kf1.v, (short)0, s[st].v, false, false);
        }

        // decay bias + causal mask
#pragma unroll
        for (int st = 0; st < 4; ++st) {
            const int j = kt * 64 + st * 16 + l15;
            const float add = rate * (float)j;
#pragma unroll
            for (int r = 0; r < 8; ++r) {
                s[st].f[r] += add;
                if (kt == qb) {
                    const int i_abs = qb * 64 + w * 16 + half * 8 + r;
                    if (j > i_abs) s[st].f[r] = -1e30f;
                }
            }
        }

        // online softmax (row reductions across the 16 column lanes)
#pragma unroll
        for (int r = 0; r < 8; ++r) {
            float mx = fmaxf(fmaxf(s[0].f[r], s[1].f[r]), fmaxf(s[2].f[r], s[3].f[r]));
#pragma unroll
            for (int d = 1; d < 16; d <<= 1) mx = fmaxf(mx, __shfl_xor(mx, d, 32));
            const float mn = fmaxf(m_i[r], mx);
            const float alpha = __expf(m_i[r] - mn);
            m_i[r] = mn;
            float rowsum = 0.0f;
#pragma unroll
            for (int st = 0; st < 4; ++st) {
                const float p = __expf(s[st].f[r] - mn);
                s[st].f[r] = p;
                rowsum += p;
            }
#pragma unroll
            for (int d = 1; d < 16; d <<= 1) rowsum += __shfl_xor(rowsum, d, 32);
            l_i[r] = l_i[r] * alpha + rowsum;
#pragma unroll
            for (int dt = 0; dt < 4; ++dt) O[dt].f[r] *= alpha;
#pragma unroll
            for (int st = 0; st < 4; ++st)
                Pl[w][half * 8 + r][st * 16 + l15] = (_Float16)s[st].f[r];
        }
        __syncthreads();

        // O += P * V
        FragA pf[2];
#pragma unroll
        for (int ks = 0; ks < 2; ++ks) {
#pragma unroll
            for (int i = 0; i < 8; ++i) {
                pf[ks].h[i]     = Pl[w][l15][ks * 32 + half * 16 + i];
                pf[ks].h[8 + i] = Pl[w][l15][ks * 32 + half * 16 + 8 + i];
            }
        }
#pragma unroll
        for (int dt = 0; dt < 4; ++dt) {
            FragA vf0, vf1;
            const int col = dt * 16 + l15;
#pragma unroll
            for (int i = 0; i < 16; ++i) {
                vf0.h[i] = Vt[col][half * 16 + i];
                vf1.h[i] = Vt[col][32 + half * 16 + i];
            }
            O[dt].v = __builtin_amdgcn_wmma_f32_16x16x32_f16(
                false, pf[0].v, false, vf0.v, (short)0, O[dt].v, false, false);
            O[dt].v = __builtin_amdgcn_wmma_f32_16x16x32_f16(
                false, pf[1].v, false, vf1.v, (short)0, O[dt].v, false, false);
        }
        __syncthreads();
    }

    // write y (heads interleaved back to [row, C])
#pragma unroll
    for (int dt = 0; dt < 4; ++dt)
#pragma unroll
        for (int r = 0; r < 8; ++r) {
            const int t_abs = qb * 64 + w * 16 + half * 8 + r;
            const float v = O[dt].f[r] / (l_i[r] + 1e-8f);
            y[((size_t)(b * TSEQ) + t_abs) * CEMB + h * HD + dt * 16 + l15] = (_Float16)v;
        }
}

// ---------------------------------------------------------------------------
// Loss: per-row max & sum-exp over vocab, then mean NLL.
// ---------------------------------------------------------------------------
__global__ __launch_bounds__(256) void rowstats_k(const float* __restrict__ logits,
                                                  float* __restrict__ stats) {
    const int row = blockIdx.x, tid = threadIdx.x;
    const float* lr = logits + (size_t)row * VOCAB;
    __shared__ float red[256];
    float mx = -1e30f;
    for (int i = tid; i < VOCAB; i += 256) mx = fmaxf(mx, lr[i]);
    red[tid] = mx; __syncthreads();
    for (int st = 128; st > 0; st >>= 1) {
        if (tid < st) red[tid] = fmaxf(red[tid], red[tid + st]);
        __syncthreads();
    }
    const float smax = red[0];
    __syncthreads();
    float se = 0.0f;
    for (int i = tid; i < VOCAB; i += 256) se += __expf(lr[i] - smax);
    red[tid] = se; __syncthreads();
    for (int st = 128; st > 0; st >>= 1) {
        if (tid < st) red[tid] += red[tid + st];
        __syncthreads();
    }
    if (tid == 0) { stats[2 * row] = smax; stats[2 * row + 1] = red[0]; }
}

__global__ __launch_bounds__(256) void loss_k(const float* __restrict__ logits,
                                              const float* __restrict__ stats,
                                              const int* __restrict__ targets,
                                              float* __restrict__ out_loss) {
    const int tid = threadIdx.x;
    __shared__ float rs[256], rc[256];
    float s = 0.0f, cnt = 0.0f;
    for (int row = tid; row < MTOT; row += 256) {
        const int tg = targets[row];
        if (tg != -1) {
            const float lv = logits[(size_t)row * VOCAB + tg];
            s += -(lv - stats[2 * row] - logf(stats[2 * row + 1]));
            cnt += 1.0f;
        }
    }
    rs[tid] = s; rc[tid] = cnt; __syncthreads();
    for (int st = 128; st > 0; st >>= 1) {
        if (tid < st) { rs[tid] += rs[tid + st]; rc[tid] += rc[tid + st]; }
        __syncthreads();
    }
    if (tid == 0) out_loss[0] = rs[0] / fmaxf(rc[0], 1.0f);
}

// ---------------------------------------------------------------------------
// Host launcher
// ---------------------------------------------------------------------------
extern "C" void kernel_launch(void* const* d_in, const int* in_sizes, int n_in,
                              void* d_out, int out_size, void* d_ws, size_t ws_size,
                              hipStream_t stream) {
    const int*   idx        = (const int*)d_in[0];
    const int*   targets    = (const int*)d_in[1];
    const float* wte        = (const float*)d_in[2];
    const float* wpe        = (const float*)d_in[3];
    const float* ln1_w      = (const float*)d_in[4];
    const float* ln1_b      = (const float*)d_in[5];
    const float* attn_w     = (const float*)d_in[6];
    const float* attn_b     = (const float*)d_in[7];
    const float* attnproj_w = (const float*)d_in[8];
    const float* attnproj_b = (const float*)d_in[9];
    const float* decay      = (const float*)d_in[10];
    const float* ln2_w      = (const float*)d_in[11];
    const float* ln2_b      = (const float*)d_in[12];
    const float* fc_w       = (const float*)d_in[13];
    const float* fc_b       = (const float*)d_in[14];
    const float* fcproj_w   = (const float*)d_in[15];
    const float* fcproj_b   = (const float*)d_in[16];
    const float* lnf_w      = (const float*)d_in[17];
    const float* lnf_b      = (const float*)d_in[18];

    float* logits = (float*)d_out;
    float* d_loss = logits + (size_t)MTOT * VOCAB;

    char* ws = (char*)d_ws;
    size_t off = 0;
    auto alloc = [&](size_t bytes) -> char* {
        char* p = ws + off;
        off += (bytes + 255) & ~(size_t)255;
        return p;
    };
    _Float16* wte16   = (_Float16*)alloc((size_t)VOCAB * CEMB * 2);
    _Float16* wqkv16  = (_Float16*)alloc((size_t)NLAYER * CEMB * 3072 * 2);
    _Float16* wproj16 = (_Float16*)alloc((size_t)NLAYER * CEMB * CEMB * 2);
    _Float16* wfc16   = (_Float16*)alloc((size_t)NLAYER * CEMB * 4096 * 2);
    _Float16* wfp16   = (_Float16*)alloc((size_t)NLAYER * 4096 * CEMB * 2);
    float*    x       = (float*)   alloc((size_t)MTOT * CEMB * 4);
    _Float16* h16     = (_Float16*)alloc((size_t)MTOT * CEMB * 2);
    _Float16* qkv16   = (_Float16*)alloc((size_t)MTOT * 3072 * 2);
    _Float16* y16     = (_Float16*)alloc((size_t)MTOT * CEMB * 2);
    _Float16* act16   = (_Float16*)alloc((size_t)MTOT * 4096 * 2);
    float*    stats   = (float*)   alloc((size_t)MTOT * 2 * 4);
    (void)n_in; (void)in_sizes; (void)out_size; (void)ws_size;

    // weight conversion
    cvt_f32_f16<<<4096, 256, 0, stream>>>(wte,        wte16,   (long long)VOCAB * CEMB);
    cvt_f32_f16<<<4096, 256, 0, stream>>>(attn_w,     wqkv16,  (long long)NLAYER * CEMB * 3072);
    cvt_f32_f16<<<4096, 256, 0, stream>>>(attnproj_w, wproj16, (long long)NLAYER * CEMB * CEMB);
    cvt_f32_f16<<<4096, 256, 0, stream>>>(fc_w,       wfc16,   (long long)NLAYER * CEMB * 4096);
    cvt_f32_f16<<<4096, 256, 0, stream>>>(fcproj_w,   wfp16,   (long long)NLAYER * 4096 * CEMB);

    embed_k<<<MTOT, 256, 0, stream>>>(idx, wte, wpe, x);

    const dim3 blk(256);
    for (int l = 0; l < NLAYER; ++l) {
        layernorm_k<<<MTOT, 256, 0, stream>>>(x, ln1_w + l * CEMB, ln1_b + l * CEMB, h16);
        // qkv = h @ Wqkv + b   -> f16
        gemm_wmma<EPI_F16, false><<<dim3(3072 / 128, MTOT / 256), blk, 0, stream>>>(
            h16, wqkv16 + (size_t)l * CEMB * 3072, attn_b + l * 3072,
            nullptr, nullptr, qkv16, MTOT, 3072, CEMB);
        attn_flash<<<dim3(TSEQ / 64, NHEAD, BB), dim3(128), 0, stream>>>(
            qkv16, decay + l * NHEAD, y16);
        // x = x + y @ Wproj + b
        gemm_wmma<EPI_RES, false><<<dim3(CEMB / 128, MTOT / 256), blk, 0, stream>>>(
            y16, wproj16 + (size_t)l * CEMB * CEMB, attnproj_b + l * CEMB,
            x, x, nullptr, MTOT, CEMB, CEMB);
        layernorm_k<<<MTOT, 256, 0, stream>>>(x, ln2_w + l * CEMB, ln2_b + l * CEMB, h16);
        // act = gelu(h @ Wfc + b) -> f16
        gemm_wmma<EPI_GELU, false><<<dim3(4096 / 128, MTOT / 256), blk, 0, stream>>>(
            h16, wfc16 + (size_t)l * CEMB * 4096, fc_b + l * 4096,
            nullptr, nullptr, act16, MTOT, 4096, CEMB);
        // x = x + act @ Wfcproj + b
        gemm_wmma<EPI_RES, false><<<dim3(CEMB / 128, MTOT / 256), blk, 0, stream>>>(
            act16, wfp16 + (size_t)l * 4096 * CEMB, fcproj_b + l * CEMB,
            x, x, nullptr, MTOT, CEMB, 4096);
    }

    layernorm_k<<<MTOT, 256, 0, stream>>>(x, lnf_w, lnf_b, h16);
    // logits = h @ wte^T  (wte is [N,K] already)
    gemm_wmma<EPI_F32, true><<<dim3(VOCAB / 128, MTOT / 256), blk, 0, stream>>>(
        h16, wte16, nullptr, nullptr, logits, nullptr, MTOT, VOCAB, CEMB);

    rowstats_k<<<MTOT, 256, 0, stream>>>(logits, stats);
    loss_k<<<1, 256, 0, stream>>>(logits, stats, targets, d_loss);
}
